// MoonElecEmb_34815004901939
// MI455X (gfx1250) — compile-verified, matrix operand order checked by reading, not run on hardware
//
#include <hip/hip_runtime.h>
#include <stdint.h>

// ---------------- problem dims ----------------
#define NE   4096
#define KN   64
#define FD   256
#define D1   32
#define D2   64
#define EENV 8
#define CUTF 5.0f
#define PI_F 3.14159265358979f

// LDS strides (in elements), padded to dodge 64-bank wrap
#define ASTR 136     // 128 K-cols + 8 pad (gamma GEMM A')
#define BSTR 136     // sBt: [256 feat][128 K + pad]
#define HSTR 260     // h_nb staging tile: 256 floats + 4 pad
#define KT   512     // kernel B reduction dim (256 pre + 256 h)
#define BSTR2 520    // 512 + 8 pad

typedef __attribute__((ext_vector_type(16))) __bf16 v16bf;
typedef __attribute__((ext_vector_type(8)))  __bf16 v8bf;
typedef __attribute__((ext_vector_type(8)))  float  v8f;
typedef __attribute__((ext_vector_type(4)))  float  v4f;
typedef __attribute__((ext_vector_type(4)))  int    v4i;

// -------- CDNA5 async global->LDS copy (ASYNCcnt path), guarded --------
#if __has_builtin(__builtin_amdgcn_global_load_async_to_lds_b128) && \
    __has_builtin(__builtin_amdgcn_s_wait_asynccnt)
#define USE_ASYNC_LDS 1
#else
#define USE_ASYNC_LDS 0
#endif

#define ASG __attribute__((address_space(1)))
#define ASL __attribute__((address_space(3)))

__device__ __forceinline__ void async_copy16(const float* gsrc, float* ldst) {
#if USE_ASYNC_LDS
    // generic->AS1: identical bit pattern; generic LDS ptr low 32 bits == LDS offset
    ASG v4i* g = (ASG v4i*)(unsigned long long)(uintptr_t)gsrc;
    ASL v4i* l = (ASL v4i*)(unsigned int)(uintptr_t)ldst;
    __builtin_amdgcn_global_load_async_to_lds_b128(g, l, 0, 0);
#else
    *(v4f*)ldst = *(const v4f*)gsrc;
#endif
}

__device__ __forceinline__ void wait_async_all() {
#if USE_ASYNC_LDS
    __builtin_amdgcn_s_wait_asynccnt(0);
#endif
}

// ---- WMMA fragment loader: 16x32 bf16 tile, row-major LDS, A-layout ----
// lane 0-15:  row = lane,    halves[0..7] = K+0..7,  halves[8..15] = K+16..23
// lane 16-31: row = lane-16, halves[0..7] = K+8..15, halves[8..15] = K+24..31
__device__ __forceinline__ v16bf load_frag(const __bf16* tile, int stride, int lane) {
    const __bf16* p = tile + (lane & 15) * stride + (((lane >> 4) & 1) << 3);
    v8bf lo = *(const v8bf*)(p);
    v8bf hi = *(const v8bf*)(p + 16);
    v16bf r;
#pragma unroll
    for (int i = 0; i < 8; ++i) { r[i] = lo[i]; r[i + 8] = hi[i]; }
    return r;
}

__device__ __forceinline__ float silu_f(float x) {
    return x / (1.0f + __expf(-x));
}

// =====================================================================
// Kernel A: one workgroup per electron n. 256 threads = 8 wave32.
//   phase0: stage weights (bf16 B' transposed) + h[n] into LDS
//   phase1: per-pair scalar pipeline -> spin-masked bf16 A' [64 x 128]
//           (async h_nb chunk 0 DMA rides underneath)
//   phase2: WMMA gamma GEMM fused with cfconv; h_nb double-buffered via
//           global_load_async_to_lds_b128, one m-tile ahead
//   phase3: write pre[n,:] (or fused VALU final projection)
// =====================================================================
__global__ void __launch_bounds__(256)
moon_pair_kernel(const float* __restrict__ r, const float* __restrict__ r_nb,
                 const float* __restrict__ h, const float* __restrict__ h_nb,
                 const int* __restrict__ s, const int* __restrict__ s_nb,
                 const float* __restrict__ ee_scales, const float* __restrict__ ee_kernel,
                 const float* __restrict__ ee_bias,
                 const float* __restrict__ W_beta2, const float* __restrict__ b_beta2,
                 const float* __restrict__ W_env,
                 const float* __restrict__ Wg_same, const float* __restrict__ Wg_diff,
                 const float* __restrict__ Wf_same, const float* __restrict__ bf_same,
                 const float* __restrict__ Wf_diff, const float* __restrict__ bf_diff,
                 const float* __restrict__ Wo, const float* __restrict__ bo,
                 const float* __restrict__ Wh,
                 float* __restrict__ pre_out, float* __restrict__ final_out,
                 int fuse_final)
{
    __shared__ __align__(16) __bf16 sA[64 * ASTR];      // spin-masked beta', A of GEMM
    __shared__ __align__(16) __bf16 sBt[FD * BSTR];     // [Wg_same;Wg_diff]^T bf16
    __shared__ __align__(16) float sHnb[2][16 * HSTR];  // h_nb async staging (double buf)
    __shared__ float x1buf[64 * D1];
    __shared__ float sWf[2 * 4 * FD];                   // [sel][c][f]
    __shared__ float sBfv[2 * FD];
    __shared__ float sWb2[D1 * D2];
    __shared__ float sBb2[D2];
    __shared__ float sWenv[EENV * D2];
    __shared__ float sEk[4 * D1];
    __shared__ float sEb[D1];
    __shared__ float sEsc[EENV];
    __shared__ float sInp[64 * 4];
    __shared__ int   sSpin[64];
    __shared__ float sH[FD];
    __shared__ float sRes[FD];

    const int n    = blockIdx.x;
    const int t    = threadIdx.x;
    const int lane = t & 31;
    const int wave = t >> 5;

    const float* hnb = h_nb + (size_t)n * KN * FD;

    // -------- phase 0: stage weights into LDS --------
    // B' = [Wg_same ; Wg_diff] stored TRANSPOSED as bf16: sBt[f][j]
    for (int i = t; i < 128 * FD; i += 256) {
        int j = i >> 8;           // K-row of B' (0..127)
        int f = i & 255;
        float v = (j < D2) ? Wg_same[j * FD + f] : Wg_diff[(j - D2) * FD + f];
        sBt[f * BSTR + j] = (__bf16)v;
    }
    for (int i = t; i < 2 * 4 * FD; i += 256) {
        int sel = i >> 10, rem = i & 1023, c = rem >> 8, f = rem & 255;
        sWf[i] = sel ? Wf_diff[c * FD + f] : Wf_same[c * FD + f];
    }
    {
        sBfv[t]       = bf_same[t & 255];       // t<256
        sBfv[FD + t]  = bf_diff[t & 255];
        for (int i = t; i < D1 * D2; i += 256) sWb2[i] = W_beta2[i];
        if (t < D2)        sBb2[t] = b_beta2[t];
        for (int i = t; i < EENV * D2; i += 256) sWenv[i] = W_env[i];
        if (t < 4 * D1)    sEk[t]  = ee_kernel[t];
        if (t < D1)        sEb[t]  = ee_bias[t];
        if (t < EENV)      sEsc[t] = ee_scales[t];
        sH[t]   = h[(size_t)n * FD + t];
        sRes[t] = 0.0f;
    }
    // prefetch the later h_nb chunks toward L2 (64 KB -> 256 x 256B lines)
    __builtin_prefetch(hnb + (size_t)t * 64, 0, 1);
#if __has_builtin(__builtin_amdgcn_s_wait_tensorcnt)
    __builtin_amdgcn_s_wait_tensorcnt(0);
#endif
    __syncthreads();

    // -------- async-stage h_nb chunk 0 (16 pairs x 256 feats = 16 KB) --------
    // rides underneath the phase-1 scalar pipeline
    {
#pragma unroll
        for (int i = 0; i < 4; ++i) {
            int idx = t + i * 256;           // 0..1023 16B packets
            int row = idx >> 6;              // pair-in-chunk 0..15
            int c4  = idx & 63;              // float4 column
            async_copy16(hnb + (size_t)row * FD + c4 * 4,
                         &sHnb[0][row * HSTR + c4 * 4]);
        }
    }

    // -------- phase 1: per-pair scalar pipeline --------
    const int k = t >> 2;    // pair 0..63
    const int q = t & 3;     // quarter

    const float rx = r[n * 3 + 0], ry = r[n * 3 + 1], rz = r[n * 3 + 2];
    const size_t pb = ((size_t)n * KN + k) * 3;
    const float dx = rx - r_nb[pb + 0];
    const float dy = ry - r_nb[pb + 1];
    const float dz = rz - r_nb[pb + 2];
    const float dist = sqrtf(dx * dx + dy * dy + dz * dz);
    const float fe0 = dist, fe1 = dx, fe2v = dy, fe3 = dz;

    // x1 = silu(feats @ ee_kernel + bias), cooperative: this thread does 8 of 32
#pragma unroll
    for (int j = 0; j < 8; ++j) {
        int d1 = q * 8 + j;
        float a = sEb[d1] + fe0 * sEk[0 * D1 + d1] + fe1 * sEk[1 * D1 + d1]
                + fe2v * sEk[2 * D1 + d1] + fe3 * sEk[3 * D1 + d1];
        x1buf[k * D1 + d1] = silu_f(a);
    }
    if (q == 0) {
        float lg  = __logf(1.0f + dist);           // log1p
        float inv = lg / dist;
        sInp[k * 4 + 0] = lg;
        sInp[k * 4 + 1] = dx * inv;
        sInp[k * 4 + 2] = dy * inv;
        sInp[k * 4 + 3] = dz * inv;
        sSpin[k] = (s[n] == s_nb[(size_t)n * KN + k]) ? 1 : 0;
    }
    __syncthreads();

    {
        float env[EENV];
#pragma unroll
        for (int e = 0; e < EENV; ++e) {
            float u = dist / sEsc[e];
            env[e] = __expf(-u * u);
        }
        float fcut = (dist < CUTF) ? 0.5f * (__cosf(PI_F * dist / CUTF) + 1.0f) : 0.0f;
        int same = sSpin[k];
        // this thread computes 16 of 64 beta components
        for (int j = 0; j < 16; ++j) {
            int d2 = q * 16 + j;
            float acc = sBb2[d2];
#pragma unroll
            for (int c = 0; c < D1; ++c) acc += x1buf[k * D1 + c] * sWb2[c * D2 + d2];
            float ew = 0.0f;
#pragma unroll
            for (int e = 0; e < EENV; ++e) ew += env[e] * sWenv[e * D2 + d2];
            float beta = acc * ew * fcut;
            // spin-masked split: cols 0-63 -> Wg_same path, 64-127 -> Wg_diff path
            sA[k * ASTR + d2 + (same ? 0  : D2)] = (__bf16)beta;
            sA[k * ASTR + d2 + (same ? D2 : 0 )] = (__bf16)0.0f;
        }
    }
    // chunk 0 complete (own counter), then barrier: sA + chunk0 visible to all
    wait_async_all();
    __syncthreads();

    // -------- phase 2: WMMA gamma GEMM + fused cfconv --------
    // 64 output tiles (4 M x 16 N of 16x16); wave w owns N-tiles {2w, 2w+1}
    v16bf bfrag[2][4];
#pragma unroll
    for (int tt = 0; tt < 2; ++tt) {
        int nt = wave * 2 + tt;
#pragma unroll
        for (int ks = 0; ks < 4; ++ks)
            bfrag[tt][ks] = load_frag(&sBt[(nt * 16) * BSTR + ks * 32], BSTR, lane);
    }

    for (int mt = 0; mt < 4; ++mt) {
        // issue async DMA for next chunk into the other buffer
        if (mt < 3) {
            float* nb = sHnb[(mt + 1) & 1];
            const float* gs = hnb + (size_t)(mt + 1) * 16 * FD;
#pragma unroll
            for (int i = 0; i < 4; ++i) {
                int idx = t + i * 256;
                int row = idx >> 6;
                int c4  = idx & 63;
                async_copy16(gs + (size_t)row * FD + c4 * 4,
                             nb + row * HSTR + c4 * 4);
            }
        }

        const float* hbuf = sHnb[mt & 1];

        v16bf afrag[4];
#pragma unroll
        for (int ks = 0; ks < 4; ++ks)
            afrag[ks] = load_frag(&sA[(mt * 16) * ASTR + ks * 32], ASTR, lane);

#pragma unroll
        for (int tt = 0; tt < 2; ++tt) {
            v8f c = {0.f, 0.f, 0.f, 0.f, 0.f, 0.f, 0.f, 0.f};
#pragma unroll
            for (int ks = 0; ks < 4; ++ks)
                c = __builtin_amdgcn_wmma_f32_16x16x32_bf16(
                        false, afrag[ks], false, bfrag[tt][ks],
                        (short)0, c, false, false);

            // cfconv: lane holds column f, local rows lr0..lr0+7 of gamma tile
            int fl  = lane & 15;
            int f   = (wave * 2 + tt) * 16 + fl;
            int lr0 = (lane >> 4) << 3;          // 0 or 8 (pair-in-chunk)
            float hn = sH[f];
            float partial = 0.0f;
#pragma unroll
            for (int v = 0; v < 8; ++v) {
                int lr  = lr0 + v;
                int kk  = mt * 16 + lr;
                int sel = sSpin[kk] ? 0 : 1;
                const float* wf = sWf + sel * (4 * FD);
                float base = sBfv[sel * FD + f];
#pragma unroll
                for (int cc = 0; cc < 4; ++cc) base += sInp[kk * 4 + cc] * wf[cc * FD + f];
                base += hn + hbuf[lr * HSTR + f];
                partial += silu_f(base) * c[v];
            }
            atomicAdd(&sRes[f], partial);   // ds_add_f32
        }

        if (mt < 3) {
            wait_async_all();    // own share of chunk mt+1 done
            __syncthreads();     // all waves' shares done; chunk mt readers done
        }
    }
    __syncthreads();

    // -------- phase 3: epilogue --------
    if (fuse_final) {
        // fallback: fused VALU GEMV final projection (only if ws too small)
        float acc = bo[t];
        for (int j = 0; j < FD; ++j)
            acc += sRes[j] * Wo[j * FD + t] + sH[j] * Wh[j * FD + t];
        final_out[(size_t)n * FD + t] = silu_f(acc);
    } else {
        pre_out[(size_t)n * FD + t] = sRes[t];
    }
}

// =====================================================================
// Kernel B: out = silu([pre, h] @ [Wo ; Wh] + bo)
//   M = 4096, K = 512, N = 256, bf16 WMMA / f32 accumulate.
//   grid (64, 4): 64-row x 64-col output blocks; 8 waves, 2 tiles each.
// =====================================================================
__global__ void __launch_bounds__(256)
moon_out_kernel(const float* __restrict__ pre, const float* __restrict__ h,
                const float* __restrict__ Wo, const float* __restrict__ bo,
                const float* __restrict__ Wh, float* __restrict__ out)
{
    __shared__ __align__(16) __bf16 sAb[64 * BSTR2];   // [m][j]  row-major
    __shared__ __align__(16) __bf16 sBb[64 * BSTR2];   // [f][j]  (B transposed)

    const int t    = threadIdx.x;
    const int lane = t & 31;
    const int wave = t >> 5;
    const int m0   = blockIdx.x * 64;
    const int f0   = blockIdx.y * 64;

    // A-tile: [64 x 512] = [pre | h], coalesced global reads
    for (int i = t; i < 64 * KT; i += 256) {
        int m = i >> 9, j = i & 511;
        float v = (j < FD) ? pre[(size_t)(m0 + m) * FD + j]
                           : h[(size_t)(m0 + m) * FD + (j - FD)];
        sAb[m * BSTR2 + j] = (__bf16)v;
    }
    // B-tile: [512 x 64] from [Wo ; Wh], stored transposed; lanes cover f
    {
        int f = t & 63;
        for (int j = (t >> 6); j < KT; j += 4) {
            float v = (j < FD) ? Wo[(size_t)j * FD + f0 + f]
                               : Wh[(size_t)(j - FD) * FD + f0 + f];
            sBb[f * BSTR2 + j] = (__bf16)v;
        }
    }
    __syncthreads();

#pragma unroll
    for (int tt = 0; tt < 2; ++tt) {
        int tid = wave * 2 + tt;
        int mt  = tid >> 2;          // 0..3
        int ntl = tid & 3;           // 0..3
        v8f c = {0.f, 0.f, 0.f, 0.f, 0.f, 0.f, 0.f, 0.f};
#pragma unroll
        for (int ks = 0; ks < 16; ++ks) {
            v16bf a = load_frag(&sAb[(mt * 16) * BSTR2 + ks * 32], BSTR2, lane);
            v16bf b = load_frag(&sBb[(ntl * 16) * BSTR2 + ks * 32], BSTR2, lane);
            c = __builtin_amdgcn_wmma_f32_16x16x32_bf16(
                    false, a, false, b, (short)0, c, false, false);
        }
        int fl    = lane & 15;
        int fg    = f0 + ntl * 16 + fl;
        int mrow0 = m0 + mt * 16 + ((lane >> 4) << 3);
        float bias = bo[fg];
#pragma unroll
        for (int v = 0; v < 8; ++v) {
            float x = c[v] + bias;
            out[(size_t)(mrow0 + v) * FD + fg] = silu_f(x);
        }
    }
}

// =====================================================================
extern "C" void kernel_launch(void* const* d_in, const int* in_sizes, int n_in,
                              void* d_out, int out_size, void* d_ws, size_t ws_size,
                              hipStream_t stream) {
    const float* r         = (const float*)d_in[0];
    const float* r_nb      = (const float*)d_in[1];
    const float* h         = (const float*)d_in[2];
    const float* h_nb      = (const float*)d_in[3];
    const int*   s         = (const int*)  d_in[4];
    const int*   s_nb      = (const int*)  d_in[5];
    const float* ee_scales = (const float*)d_in[6];
    const float* ee_kernel = (const float*)d_in[7];
    const float* ee_bias   = (const float*)d_in[8];
    const float* W_beta2   = (const float*)d_in[9];
    const float* b_beta2   = (const float*)d_in[10];
    const float* W_env     = (const float*)d_in[11];
    const float* Wg_same   = (const float*)d_in[12];
    const float* Wg_diff   = (const float*)d_in[13];
    const float* Wf_same   = (const float*)d_in[14];
    const float* bf_same   = (const float*)d_in[15];
    const float* Wf_diff   = (const float*)d_in[16];
    const float* bf_diff   = (const float*)d_in[17];
    const float* Wo        = (const float*)d_in[18];
    const float* bo        = (const float*)d_in[19];
    const float* Wh        = (const float*)d_in[20];
    float* out = (float*)d_out;

    const size_t pre_bytes = (size_t)NE * FD * sizeof(float);
    const int fuse = (ws_size < pre_bytes) ? 1 : 0;   // deterministic per-config
    float* pre = (float*)d_ws;

    moon_pair_kernel<<<NE, 256, 0, stream>>>(
        r, r_nb, h, h_nb, s, s_nb, ee_scales, ee_kernel, ee_bias,
        W_beta2, b_beta2, W_env, Wg_same, Wg_diff,
        Wf_same, bf_same, Wf_diff, bf_diff, Wo, bo, Wh,
        pre, out, fuse);

    if (!fuse) {
        moon_out_kernel<<<dim3(NE / 64, FD / 64), 256, 0, stream>>>(
            pre, h, Wo, bo, Wh, out);
    }
}
